// DrQA_31018253811717
// MI455X (gfx1250) — matrix operand order repaired and checked by reading, not compiled, not run on hardware
//
#include <hip/hip_runtime.h>
#include <hip/hip_bf16.h>

typedef __attribute__((ext_vector_type(16))) _Float16 v16h;
typedef __attribute__((ext_vector_type(8)))  float    v8f;

#define B_   64
#define P_   512
#define Q_   32
#define D_   300
#define H_   128
#define G4_  512          // 4*H
#define PIN_PAD 672       // 671 -> 672 (21 k-chunks of 32)
#define QIN_PAD 320       // 300 -> 320 (10 k-chunks of 32)

__device__ __forceinline__ float sigmoidf_(float x) { return 1.0f / (1.0f + expf(-x)); }

// CDNA5 async copy global -> LDS (16B per lane), tracked by ASYNCcnt.
__device__ __forceinline__ void async_b128(uint32_t lds_byte_addr, const void* gaddr) {
    asm volatile("global_load_async_to_lds_b128 %0, %1, off"
                 :: "v"(lds_byte_addr), "v"(gaddr) : "memory");
}
#define WAIT_ASYNC(n) asm volatile("s_wait_asynccnt " #n ::: "memory")

// On gfx1250 a flat shared-aperture address carries the LDS byte offset in its
// low 32 bits (ISA 10.2).
__device__ __forceinline__ uint32_t lds_u32(const void* p) {
    return (uint32_t)(uintptr_t)p;
}

// ---------------------------------------------------------------------------
// Convert fp32 weight [rows, ks] -> f16 [rows, kd] with zero pad on K.
// ---------------------------------------------------------------------------
__global__ void cvt_pad(const float* __restrict__ src, _Float16* __restrict__ dst,
                        int rows, int ks, int kd) {
    int idx = blockIdx.x * blockDim.x + threadIdx.x;
    if (idx >= rows * kd) return;
    int r = idx / kd, k = idx - r * kd;
    float v = (k < ks) ? src[(size_t)r * ks + k] : 0.0f;
    dst[idx] = (_Float16)v;
}

__global__ void add_bias2(const float* __restrict__ a, const float* __restrict__ b,
                          float* __restrict__ o, int n) {
    int i = blockIdx.x * blockDim.x + threadIdx.x;
    if (i < n) o[i] = a[i] + b[i];
}

// ---------------------------------------------------------------------------
// Per-batch query features: qa = relu(qemb @ w_alpha + b_alpha),
// aligned[b,:] = sum_q (qa/sum qa) * qemb   (paragraph factor pa cancels)
// also emits f16 query embedding [B*Q, 320] for the q-LSTM input GEMM.
// ---------------------------------------------------------------------------
__global__ void query_feats(const int* __restrict__ query, const float* __restrict__ emb,
                            const float* __restrict__ w_alpha, const float* __restrict__ b_alpha,
                            float* __restrict__ aligned, _Float16* __restrict__ qembf16) {
    int b = blockIdx.x;
    __shared__ float qe[Q_][D_ + 4];
    __shared__ float qa[Q_];
    __shared__ float Ssum;
    for (int i = threadIdx.x; i < Q_ * D_; i += blockDim.x) {
        int q = i / D_, d = i - q * D_;
        float v = emb[(size_t)query[b * Q_ + q] * D_ + d];
        qe[q][d] = v;
        qembf16[((size_t)b * Q_ + q) * QIN_PAD + d] = (_Float16)v;
    }
    for (int i = threadIdx.x; i < Q_ * (QIN_PAD - D_); i += blockDim.x) {
        int q = i / (QIN_PAD - D_), d = D_ + (i - q * (QIN_PAD - D_));
        qembf16[((size_t)b * Q_ + q) * QIN_PAD + d] = (_Float16)0.0f;
    }
    __syncthreads();
    if (threadIdx.x < Q_) {
        int q = threadIdx.x;
        float acc = b_alpha[0];
        for (int d = 0; d < D_; ++d) acc += qe[q][d] * w_alpha[d];
        qa[q] = fmaxf(acc, 0.0f);
    }
    __syncthreads();
    if (threadIdx.x == 0) {
        float s = 0.0f;
        for (int q = 0; q < Q_; ++q) s += qa[q];
        Ssum = s;
    }
    __syncthreads();
    for (int d = threadIdx.x; d < D_; d += blockDim.x) {
        float acc = 0.0f;
        for (int q = 0; q < Q_; ++q) acc += qa[q] * qe[q][d];
        aligned[b * D_ + d] = acc / Ssum;
    }
}

// ---------------------------------------------------------------------------
// Build paragraph feature rows (f16): [emb(300) | ner_oh(20) | pos_oh(50) |
// aligned(300) | match(1) | pad(1)] -> 672 wide. One block per token.
// ---------------------------------------------------------------------------
__global__ void build_conc(const int* __restrict__ pars, const int* __restrict__ query,
                           const int* __restrict__ ind2ner, const int* __restrict__ ind2pos,
                           const float* __restrict__ emb, const float* __restrict__ aligned,
                           _Float16* __restrict__ conc) {
    int tok = blockIdx.x;                 // 0..32767
    int b = tok >> 9;
    int w = pars[tok];
    __shared__ int matchS;
    if (threadIdx.x == 0) {
        int m = 0;
        for (int q = 0; q < Q_; ++q) m |= (query[b * Q_ + q] == w) ? 1 : 0;
        matchS = m;
    }
    __syncthreads();
    int ner = ind2ner[w], pos = ind2pos[w];
    const float* e  = emb + (size_t)w * D_;
    const float* al = aligned + b * D_;
    _Float16* out = conc + (size_t)tok * PIN_PAD;
    for (int k = threadIdx.x; k < PIN_PAD; k += blockDim.x) {
        float v;
        if      (k < 300) v = e[k];
        else if (k < 320) v = (k - 300 == ner) ? 1.0f : 0.0f;
        else if (k < 370) v = (k - 320 == pos) ? 1.0f : 0.0f;
        else if (k < 670) v = al[k - 370];
        else if (k == 670) v = (float)matchS;
        else v = 0.0f;
        out[k] = (_Float16)v;
    }
}

// ---------------------------------------------------------------------------
// Tiled WMMA GEMM with async global->LDS double buffering.
//   C[M,N] = A[M,K](f16,row-major,lda) @ W[N,K]^T (f16,row-major,ldb) + bias[N]
// Block tile 128x64: 8 waves, wave w = M-tile w x 4 N-tiles.
// Per k-chunk (K=32): A-tile 128x32 (8KB) + B-tile 64x32 (4KB) staged by 3
// async_to_lds_b128 per thread; next chunk's DMA overlaps current WMMAs.
// Fragment layouts per ISA: A interleaved 8-half runs per half-wave,
// B interleaved 8-half runs per half-wave, C VGPR r <-> row r+8*half.
// ---------------------------------------------------------------------------
__global__ void wmma_gemm_tiled(const _Float16* __restrict__ A, int lda,
                                const _Float16* __restrict__ W, int ldb,
                                const float* __restrict__ bias,
                                float* __restrict__ C, int ldc,
                                int N, int K) {
    __shared__ _Float16 As[2][128 * 32];   // 2 x 8KB
    __shared__ _Float16 Bs[2][64 * 32];    // 2 x 4KB
    int nBlk = N >> 6;
    int mb = blockIdx.x / nBlk, nb = blockIdx.x - mb * nBlk;
    int tid  = threadIdx.x;
    int wave = tid >> 5, lane = tid & 31, half = lane >> 4, l16 = lane & 15;

    // staging assignment: thread t moves A 16B-chunks t and t+256, B chunk t
    int ar0 = tid >> 2, sub = tid & 3;
    int ar1 = ar0 + 64;
    const _Float16* Ag0 = A + (size_t)(mb * 128 + ar0) * lda + sub * 8;
    const _Float16* Ag1 = A + (size_t)(mb * 128 + ar1) * lda + sub * 8;
    const _Float16* Bg  = W + (size_t)(nb * 64  + ar0) * ldb + sub * 8;
    uint32_t Al0[2], Al1[2], Bl[2];
#pragma unroll
    for (int bu = 0; bu < 2; ++bu) {
        Al0[bu] = lds_u32(&As[bu][ar0 * 32 + sub * 8]);
        Al1[bu] = lds_u32(&As[bu][ar1 * 32 + sub * 8]);
        Bl[bu]  = lds_u32(&Bs[bu][ar0 * 32 + sub * 8]);
    }

    v8f acc[4] = {v8f{}, v8f{}, v8f{}, v8f{}};
    int nk = K >> 5;

    // prefetch chunk 0
    async_b128(Al0[0], Ag0);
    async_b128(Al1[0], Ag1);
    async_b128(Bl[0],  Bg);

    for (int i = 0; i < nk; ++i) {
        int cur = i & 1;
        if (i + 1 < nk) {                       // prefetch next chunk
            int ko = (i + 1) << 5;
            async_b128(Al0[cur ^ 1], Ag0 + ko);
            async_b128(Al1[cur ^ 1], Ag1 + ko);
            async_b128(Bl[cur ^ 1],  Bg  + ko);
            WAIT_ASYNC(3);                      // drain chunk i, keep i+1 in flight
        } else {
            WAIT_ASYNC(0);
        }
        __syncthreads();

        const _Float16* ap = &As[cur][(wave * 16 + l16) * 32 + 8 * half];
        v16h a;
        *((float4*)&a)     = *(const float4*)(ap);
        *((float4*)&a + 1) = *(const float4*)(ap + 16);
#pragma unroll
        for (int j = 0; j < 4; ++j) {
            const _Float16* bp = &Bs[cur][(j * 16 + l16) * 32 + 8 * half];
            v16h bf;
            *((float4*)&bf)     = *(const float4*)(bp);
            *((float4*)&bf + 1) = *(const float4*)(bp + 16);
            acc[j] = __builtin_amdgcn_wmma_f32_16x16x32_f16(
                false, a, false, bf, (short)0, acc[j], false, false);
        }
        __syncthreads();                        // protect buffer before re-stage
    }

#pragma unroll
    for (int j = 0; j < 4; ++j) {
        int col = nb * 64 + j * 16 + l16;
        float bv = bias[col];
#pragma unroll
        for (int r = 0; r < 8; ++r) {
            int rr = mb * 128 + wave * 16 + r + 8 * half;
            C[(size_t)rr * ldc + col] = acc[j][r] + bv;
        }
    }
}

// ---------------------------------------------------------------------------
// LSTM recurrence. One block per (lstm, dir): blockIdx 0=p-fwd,1=p-bwd,
// 2=q-fwd,3=q-bwd. 256 threads = 8 waves; wave w owns hidden cols [16w,16w+16).
// h kept as f16 in double-buffered LDS (WMMA A operand); c stays in fp32 VGPRs.
// Per step: 4 M-tiles x 4 gate-tiles x 4 k-chunks = 64 WMMAs/wave, 1 barrier.
// ---------------------------------------------------------------------------
__global__ void lstm_rec(const float* __restrict__ xg_p, const float* __restrict__ xg_q,
                         const _Float16* __restrict__ whhP, const _Float16* __restrict__ whhQ,
                         float* __restrict__ res) {
    int id  = blockIdx.x;
    int isQ = id >> 1;
    int dir = id & 1;
    int T = isQ ? Q_ : P_;
    const float*     xg  = (isQ ? xg_q : xg_p) + (size_t)dir * (size_t)B_ * T * G4_;
    const _Float16*  whh = (isQ ? whhQ : whhP) + (size_t)dir * G4_ * H_;
    int hOff = isQ * 512 + dir * 128;
    int cOff = hOff + 256;

    __shared__ _Float16 hbuf[2][B_ * H_];      // 32 KB
    for (int i = threadIdx.x; i < B_ * H_; i += blockDim.x) hbuf[0][i] = (_Float16)0.0f;
    __syncthreads();

    int wave = threadIdx.x >> 5, lane = threadIdx.x & 31;
    int half = lane >> 4, l16 = lane & 15;
    int col  = wave * 16 + l16;                 // hidden column of this lane's C elems
    v8f cReg[4] = {v8f{}, v8f{}, v8f{}, v8f{}};

    for (int s = 0; s < T; ++s) {
        int t   = dir ? (T - 1 - s) : s;
        int cur = s & 1, nxt = cur ^ 1;
#pragma unroll
        for (int mt = 0; mt < 4; ++mt) {
            v8f acc[4];
#pragma unroll
            for (int g = 0; g < 4; ++g)
#pragma unroll
                for (int r = 0; r < 8; ++r) {
                    int bb = mt * 16 + r + 8 * half;
                    acc[g][r] = xg[((size_t)bb * T + t) * G4_ + g * 128 + col];
                }
#pragma unroll
            for (int kb = 0; kb < H_; kb += 32) {
                v16h a;
                const _Float16* ap = &hbuf[cur][(mt * 16 + l16) * H_ + kb + 8 * half];
                *((float4*)&a)     = *(const float4*)(ap);
                *((float4*)&a + 1) = *(const float4*)(ap + 16);
#pragma unroll
                for (int g = 0; g < 4; ++g) {
                    const _Float16* bp = whh + (size_t)(g * 128 + col) * H_ + kb + 8 * half;
                    v16h bf;
                    *((float4*)&bf)     = *(const float4*)(bp);
                    *((float4*)&bf + 1) = *(const float4*)(bp + 16);
                    acc[g] = __builtin_amdgcn_wmma_f32_16x16x32_f16(
                        false, a, false, bf, (short)0, acc[g], false, false);
                }
            }
#pragma unroll
            for (int r = 0; r < 8; ++r) {
                int bb = mt * 16 + r + 8 * half;
                float iv = sigmoidf_(acc[0][r]);
                float fv = sigmoidf_(acc[1][r]);
                float gv = tanhf(acc[2][r]);
                float ov = sigmoidf_(acc[3][r]);
                float cN = fv * cReg[mt][r] + iv * gv;
                float hN = ov * tanhf(cN);
                cReg[mt][r] = cN;
                hbuf[nxt][bb * H_ + col] = (_Float16)hN;
                if (s == T - 1) {
                    res[(size_t)bb * 1024 + hOff + col] = hN;
                    res[(size_t)bb * 1024 + cOff + col] = cN;
                }
            }
        }
        __syncthreads();
    }
}

// ---------------------------------------------------------------------------
// FC head: out = (res @ fc1^T + b1) @ fc2^T + b2.  Tiny; exact fp32.
// ---------------------------------------------------------------------------
__global__ void fc_head(const float* __restrict__ res,
                        const float* __restrict__ fc1w, const float* __restrict__ fc1b,
                        const float* __restrict__ fc2w, const float* __restrict__ fc2b,
                        float* __restrict__ out) {
    int b = blockIdx.x;
    __shared__ float r[1024];
    __shared__ float h1[256];
    for (int i = threadIdx.x; i < 1024; i += blockDim.x) r[i] = res[(size_t)b * 1024 + i];
    __syncthreads();
    int j = threadIdx.x;
    float acc = fc1b[j];
    for (int k = 0; k < 1024; ++k) acc += fc1w[(size_t)j * 1024 + k] * r[k];
    h1[j] = acc;
    __syncthreads();
    if (j < 2) {
        float a2 = fc2b[j];
        for (int k = 0; k < 256; ++k) a2 += fc2w[j * 256 + k] * h1[k];
        out[b * 2 + j] = a2;
    }
}

// ---------------------------------------------------------------------------
extern "C" void kernel_launch(void* const* d_in, const int* in_sizes, int n_in,
                              void* d_out, int out_size, void* d_ws, size_t ws_size,
                              hipStream_t stream) {
    (void)in_sizes; (void)n_in; (void)out_size; (void)ws_size;
    const int*   pars    = (const int*)d_in[0];
    const int*   query   = (const int*)d_in[1];
    const int*   ind2ner = (const int*)d_in[2];
    const int*   ind2pos = (const int*)d_in[3];
    const float* emb     = (const float*)d_in[4];
    const float* w_alpha = (const float*)d_in[5];
    const float* b_alpha = (const float*)d_in[6];
    const float* pWih[2] = {(const float*)d_in[7],  (const float*)d_in[11]};
    const float* pWhh[2] = {(const float*)d_in[8],  (const float*)d_in[12]};
    const float* pbih[2] = {(const float*)d_in[9],  (const float*)d_in[13]};
    const float* pbhh[2] = {(const float*)d_in[10], (const float*)d_in[14]};
    const float* qWih[2] = {(const float*)d_in[15], (const float*)d_in[19]};
    const float* qWhh[2] = {(const float*)d_in[16], (const float*)d_in[20]};
    const float* qbih[2] = {(const float*)d_in[17], (const float*)d_in[21]};
    const float* qbhh[2] = {(const float*)d_in[18], (const float*)d_in[22]};
    const float* fc1w = (const float*)d_in[23];
    const float* fc1b = (const float*)d_in[24];
    const float* fc2w = (const float*)d_in[25];
    const float* fc2b = (const float*)d_in[26];
    float* out = (float*)d_out;

    char* wp = (char*)d_ws;
    auto carve = [&](size_t bytes) -> char* {
        char* r = wp; wp += (bytes + 255) & ~(size_t)255; return r;
    };
    _Float16* conc    = (_Float16*)carve((size_t)B_ * P_ * PIN_PAD * 2);
    _Float16* qembf16 = (_Float16*)carve((size_t)B_ * Q_ * QIN_PAD * 2);
    float*    aligned = (float*)   carve((size_t)B_ * D_ * 4);
    _Float16* wP      = (_Float16*)carve((size_t)2 * G4_ * PIN_PAD * 2);
    _Float16* wQ      = (_Float16*)carve((size_t)2 * G4_ * QIN_PAD * 2);
    _Float16* whhP    = (_Float16*)carve((size_t)2 * G4_ * H_ * 2);
    _Float16* whhQ    = (_Float16*)carve((size_t)2 * G4_ * H_ * 2);
    float*    biasP   = (float*)   carve((size_t)2 * G4_ * 4);
    float*    biasQ   = (float*)   carve((size_t)2 * G4_ * 4);
    float*    xg_p    = (float*)   carve((size_t)2 * B_ * P_ * G4_ * 4);
    float*    xg_q    = (float*)   carve((size_t)2 * B_ * Q_ * G4_ * 4);
    float*    res     = (float*)   carve((size_t)B_ * 1024 * 4);

    // 1) weight conversion / bias fold
    for (int d = 0; d < 2; ++d) {
        int n;
        n = G4_ * PIN_PAD;
        cvt_pad<<<(n + 255) / 256, 256, 0, stream>>>(pWih[d], wP + (size_t)d * n, G4_, 671, PIN_PAD);
        n = G4_ * QIN_PAD;
        cvt_pad<<<(n + 255) / 256, 256, 0, stream>>>(qWih[d], wQ + (size_t)d * n, G4_, 300, QIN_PAD);
        n = G4_ * H_;
        cvt_pad<<<(n + 255) / 256, 256, 0, stream>>>(pWhh[d], whhP + (size_t)d * n, G4_, H_, H_);
        cvt_pad<<<(n + 255) / 256, 256, 0, stream>>>(qWhh[d], whhQ + (size_t)d * n, G4_, H_, H_);
        add_bias2<<<2, 256, 0, stream>>>(pbih[d], pbhh[d], biasP + d * G4_, G4_);
        add_bias2<<<2, 256, 0, stream>>>(qbih[d], qbhh[d], biasQ + d * G4_, G4_);
    }

    // 2) query features (qa, aligned, f16 query embedding)
    query_feats<<<B_, 128, 0, stream>>>(query, emb, w_alpha, b_alpha, aligned, qembf16);

    // 3) paragraph feature rows (f16)
    build_conc<<<B_ * P_, 128, 0, stream>>>(pars, query, ind2ner, ind2pos, emb, aligned, conc);

    // 4) input-gate GEMMs (tiled WMMA + async LDS staging)
    {
        int Mp = B_ * P_;                                   // 32768
        int blocksP = (Mp / 128) * (G4_ / 64);              // 2048
        for (int d = 0; d < 2; ++d)
            wmma_gemm_tiled<<<blocksP, 256, 0, stream>>>(
                conc, PIN_PAD, wP + (size_t)d * G4_ * PIN_PAD, PIN_PAD,
                biasP + d * G4_, xg_p + (size_t)d * Mp * G4_, G4_, G4_, PIN_PAD);
        int Mq = B_ * Q_;                                   // 2048
        int blocksQ = (Mq / 128) * (G4_ / 64);              // 128
        for (int d = 0; d < 2; ++d)
            wmma_gemm_tiled<<<blocksQ, 256, 0, stream>>>(
                qembf16, QIN_PAD, wQ + (size_t)d * G4_ * QIN_PAD, QIN_PAD,
                biasQ + d * G4_, xg_q + (size_t)d * Mq * G4_, G4_, G4_, QIN_PAD);
    }

    // 5) recurrences: p-fwd, p-bwd, q-fwd, q-bwd in parallel (4 blocks)
    lstm_rec<<<4, 256, 0, stream>>>(xg_p, xg_q, whhP, whhQ, res);

    // 6) FC head
    fc_head<<<B_, 256, 0, stream>>>(res, fc1w, fc1b, fc2w, fc2b, out);
}
